// Transformer_33689723470026
// MI455X (gfx1250) — matrix operand order repaired and checked by reading, not compiled, not run on hardware
//
#include <hip/hip_runtime.h>
#include <stdint.h>

// ---------------------------------------------------------------------------
// Types for CDNA5 WMMA (wave32): bf16 A/B fragments (8 VGPRs -> v16bf), f32 C/D
// ---------------------------------------------------------------------------
typedef __bf16 bf16;
typedef __attribute__((ext_vector_type(16))) __bf16 v16bf;
typedef __attribute__((ext_vector_type(8)))  __bf16 v8bf;
typedef __attribute__((ext_vector_type(8)))  float  v8f;
typedef __attribute__((ext_vector_type(4)))  int    v4i;

__device__ __forceinline__ bf16 tobf(float f) { return (bf16)f; }

// ---------------------------------------------------------------------------
// Async global->LDS copy (CDNA5 ASYNCcnt path), with register-copy fallback.
// Builtin takes (global int4*, local int4*, i32 offset, i32 cpol).
// ---------------------------------------------------------------------------
#if defined(__has_builtin)
#  if __has_builtin(__builtin_amdgcn_global_load_async_to_lds_b128) && \
      __has_builtin(__builtin_amdgcn_s_wait_asynccnt)
#    define HAVE_ASYNC_LDS 1
#  endif
#endif
#ifndef HAVE_ASYNC_LDS
#  define HAVE_ASYNC_LDS 0
#endif

typedef __attribute__((address_space(1))) v4i* gptr_v4i;
typedef __attribute__((address_space(3))) v4i* lptr_v4i;

__device__ __forceinline__ void cp16_g2l(bf16* dst_lds, const bf16* src_glob) {
#if HAVE_ASYNC_LDS
    __builtin_amdgcn_global_load_async_to_lds_b128(
        (gptr_v4i)(src_glob), (lptr_v4i)(dst_lds), 0, 0);
#else
    *(v8bf*)dst_lds = *(const v8bf*)src_glob;
#endif
}
__device__ __forceinline__ void cp_wait() {
#if HAVE_ASYNC_LDS
    __builtin_amdgcn_s_wait_asynccnt(0);
#endif
}

// ---------------------------------------------------------------------------
// bf16-in WMMA GEMM:  C[z] = alpha * A[z] x B[z](^T) + bias[z]  (opt relu)
// Block tile 128(M) x 64(N), K-step 64, DOUBLE-BUFFERED LDS: async DMA of
// stage s+1 overlaps the 8 v_wmma_f32_16x16x32_bf16 of stage s.
// All dims divide tiles exactly in this model (no edge handling).
// Output: f32 (Cf) or bf16 (Cbf) -- exactly one non-null.
// ---------------------------------------------------------------------------
#define BM 128
#define BN 64
#define BK 64
#define AST 72   // LDS row stride (bf16) for A tile (64 + 8 pad)
#define BST 72   // LDS row stride (bf16) for K-transposed B tile

__global__ __launch_bounds__(256)
void gemm_wmma_k(const bf16* __restrict__ Ab, const bf16* __restrict__ Bb,
                 const float* __restrict__ biasb,
                 float* __restrict__ Cfb, bf16* __restrict__ Cbfb,
                 int K, int lda, int ldb, int ldc,
                 long long sA, long long sB, long long sC, long long sBias,
                 float alpha, int transB, int relu, int causal)
{
    const int z = blockIdx.z;
    const bf16* A = Ab + (long long)z * sA;
    const bf16* B = Bb + (long long)z * sB;
    const float* bias = biasb ? (biasb + (long long)z * sBias) : nullptr;
    float* Cf  = Cfb  ? (Cfb  + (long long)z * sC) : nullptr;
    bf16*  Cbf = Cbfb ? (Cbfb + (long long)z * sC) : nullptr;

    const int rowBase = blockIdx.y * BM;
    const int colBase = blockIdx.x * BN;
    if (causal && colBase > rowBase + (BM - 1)) return;   // fully-masked block

    __shared__ __align__(16) bf16 As[2][BM * AST];        // [m][k], double buffered
    __shared__ __align__(16) bf16 Bs[2][BN * BST];        // [n][k] (K-transposed)

    const int tid    = threadIdx.x;
    const int lane   = tid & 31;
    const int wave   = tid >> 5;
    const int wm     = (wave & 3) * 32;                   // 4 x 32 = 128 (M)
    const int wn     = (wave >> 2) * 32;                  // 2 x 32 = 64  (N)
    const int laneHi = lane >> 4;
    const int l15    = lane & 15;
    const int kA0    = laneHi ? 8 : 0;                    // A frag chunks {kA0, kA0+16}
    const int kB0    = laneHi ? 8 : 0;                    // B frag chunks {kB0, kB0+16}

    // staging helpers -------------------------------------------------------
    auto issueA = [&](int k0, int buf) {                  // 128x64 bf16, 4x16B/thread
        #pragma unroll
        for (int i = 0; i < 4; ++i) {
            const int j   = tid + 256 * i;
            const int r   = j >> 3;                       // row 0..127
            const int kk8 = (j & 7) << 3;                 // k 0,8,..,56
            cp16_g2l(&As[buf][r * AST + kk8],
                     A + (long long)(rowBase + r) * lda + k0 + kk8);
        }
    };
    auto issueBtrans = [&](int k0, int buf) {             // B[N][K]: pure async copy
        #pragma unroll
        for (int i = 0; i < 2; ++i) {
            const int j   = tid + 256 * i;
            const int nn  = j >> 3;                       // 0..63
            const int kk8 = (j & 7) << 3;                 // 0,8,..,56
            cp16_g2l(&Bs[buf][nn * BST + kk8],
                     B + (long long)(colBase + nn) * ldb + k0 + kk8);
        }
    };
    const int kr0 = tid >> 3;                             // 0..31 (non-trans B path)
    const int nn8 = (tid & 7) << 3;                       // 0,8,..,56
    auto loadBn = [&](int k0, v8bf& t0, v8bf& t1) {       // B[K][N]: global loads
        t0 = *(const v8bf*)(B + (long long)(k0 + kr0)      * ldb + colBase + nn8);
        t1 = *(const v8bf*)(B + (long long)(k0 + kr0 + 32) * ldb + colBase + nn8);
    };
    auto storeBn = [&](int buf, v8bf t0, v8bf t1) {       // transpose scatter to LDS
        #pragma unroll
        for (int e = 0; e < 8; ++e) {
            Bs[buf][(nn8 + e) * BST + kr0]      = t0[e];
            Bs[buf][(nn8 + e) * BST + kr0 + 32] = t1[e];
        }
    };

    v8f c00 = {0,0,0,0,0,0,0,0};
    v8f c01 = {0,0,0,0,0,0,0,0};
    v8f c10 = {0,0,0,0,0,0,0,0};
    v8f c11 = {0,0,0,0,0,0,0,0};

    // prologue: stage 0 into buffer 0 --------------------------------------
    issueA(0, 0);
    if (transB) {
        issueBtrans(0, 0);
    } else {
        v8bf p0, p1;
        loadBn(0, p0, p1);
        storeBn(0, p0, p1);
    }
    cp_wait();
    __syncthreads();

    const int nStages = K / BK;
    for (int s = 0; s < nStages; ++s) {
        const int buf = s & 1;
        const bool pre = (s + 1 < nStages);

        // ---- issue stage s+1 copies (DMA overlaps this stage's WMMAs) ----
        v8bf t0, t1;
        if (pre) {
            const int k0n = (s + 1) * BK;
            issueA(k0n, buf ^ 1);
            if (transB) issueBtrans(k0n, buf ^ 1);
            else        loadBn(k0n, t0, t1);              // loads in flight
        }

        // ---- 2 WMMA K-steps of 32 from buffer `buf` ----
        const bf16* Asb = As[buf];
        const bf16* Bsb = Bs[buf];
        #pragma unroll
        for (int ks = 0; ks < BK; ks += 32) {
            const int ar0 = wm + l15;
            const int ar1 = wm + 16 + l15;
            v8bf a0lo = *(const v8bf*)&Asb[ar0 * AST + ks + kA0];
            v8bf a0hi = *(const v8bf*)&Asb[ar0 * AST + ks + kA0 + 16];
            v8bf a1lo = *(const v8bf*)&Asb[ar1 * AST + ks + kA0];
            v8bf a1hi = *(const v8bf*)&Asb[ar1 * AST + ks + kA0 + 16];
            v16bf a0 = __builtin_shufflevector(a0lo, a0hi, 0,1,2,3,4,5,6,7,8,9,10,11,12,13,14,15);
            v16bf a1 = __builtin_shufflevector(a1lo, a1hi, 0,1,2,3,4,5,6,7,8,9,10,11,12,13,14,15);

            const int bc0 = wn + l15;
            const int bc1 = wn + 16 + l15;
            v8bf b0lo = *(const v8bf*)&Bsb[bc0 * BST + ks + kB0];
            v8bf b0hi = *(const v8bf*)&Bsb[bc0 * BST + ks + kB0 + 16];
            v8bf b1lo = *(const v8bf*)&Bsb[bc1 * BST + ks + kB0];
            v8bf b1hi = *(const v8bf*)&Bsb[bc1 * BST + ks + kB0 + 16];
            v16bf b0 = __builtin_shufflevector(b0lo, b0hi, 0,1,2,3,4,5,6,7,8,9,10,11,12,13,14,15);
            v16bf b1 = __builtin_shufflevector(b1lo, b1hi, 0,1,2,3,4,5,6,7,8,9,10,11,12,13,14,15);

            c00 = __builtin_amdgcn_wmma_f32_16x16x32_bf16(false, a0, false, b0, (short)0, c00, false, false);
            c01 = __builtin_amdgcn_wmma_f32_16x16x32_bf16(false, a0, false, b1, (short)0, c01, false, false);
            c10 = __builtin_amdgcn_wmma_f32_16x16x32_bf16(false, a1, false, b0, (short)0, c10, false, false);
            c11 = __builtin_amdgcn_wmma_f32_16x16x32_bf16(false, a1, false, b1, (short)0, c11, false, false);
        }

        // ---- commit deferred B transpose, wait DMA, one barrier/stage ----
        if (pre) {
            if (!transB) storeBn(buf ^ 1, t0, t1);        // loadcnt wait lands here
            cp_wait();
        }
        __syncthreads();
    }

    // ---- epilogue: f32 C/D lane layout (lanes 0-15: M=r, 16-31: M=8+r) ----
    #pragma unroll
    for (int sm = 0; sm < 2; ++sm) {
        #pragma unroll
        for (int sn = 0; sn < 2; ++sn) {
            v8f acc = sm == 0 ? (sn == 0 ? c00 : c01) : (sn == 0 ? c10 : c11);
            const int ncol = colBase + wn + sn * 16 + l15;
            const float bv = bias ? bias[ncol] : 0.0f;
            #pragma unroll
            for (int r = 0; r < 8; ++r) {
                const int mrow = rowBase + wm + sm * 16 + (laneHi ? 8 + r : r);
                float val = acc[r] * alpha + bv;
                if (relu) val = fmaxf(val, 0.0f);
                const long long idx = (long long)mrow * ldc + ncol;
                if (Cf)  Cf[idx]  = val;
                if (Cbf) Cbf[idx] = tobf(val);
            }
        }
    }
}

// ---------------------------------------------------------------------------
// Streaming f32 -> bf16 conversion (weights/activations), 8 elems per thread
// ---------------------------------------------------------------------------
__global__ __launch_bounds__(256)
void cvt_bf16_k(const float* __restrict__ src, bf16* __restrict__ dst)
{
    const long long i = ((long long)blockIdx.x * 256 + threadIdx.x) * 8;
    const float4 a = *(const float4*)(src + i);
    const float4 b = *(const float4*)(src + i + 4);
    v8bf o = { tobf(a.x), tobf(a.y), tobf(a.z), tobf(a.w),
               tobf(b.x), tobf(b.y), tobf(b.z), tobf(b.w) };
    *(v8bf*)(dst + i) = o;
}

// ---------------------------------------------------------------------------
// Elementwise / reduction kernels
// ---------------------------------------------------------------------------
__global__ __launch_bounds__(256)
void embed_k(const int* __restrict__ inp, const float* __restrict__ tok,
             const float* __restrict__ pos, float* __restrict__ x, int D)
{
    const long long i = (long long)blockIdx.x * 256 + threadIdx.x;
    const int t = (int)(i / D);
    const int d = (int)(i % D);
    x[i] = tok[(long long)inp[t] * D + d] + pos[i];
}

__global__ __launch_bounds__(256)
void ln_k(const float* __restrict__ x, const float* __restrict__ g,
          const float* __restrict__ b, bf16* __restrict__ y, int D)
{
    const int t = blockIdx.x;
    const float* xr = x + (long long)t * D;
    bf16* yr = y + (long long)t * D;
    const int tid = threadIdx.x;
    __shared__ float red[256];

    float s = 0.0f;
    for (int i = tid; i < D; i += 256) s += xr[i];
    red[tid] = s; __syncthreads();
    for (int o = 128; o > 0; o >>= 1) { if (tid < o) red[tid] += red[tid + o]; __syncthreads(); }
    const float mean = red[0] / (float)D;
    __syncthreads();

    float v = 0.0f;
    for (int i = tid; i < D; i += 256) { const float d = xr[i] - mean; v += d * d; }
    red[tid] = v; __syncthreads();
    for (int o = 128; o > 0; o >>= 1) { if (tid < o) red[tid] += red[tid + o]; __syncthreads(); }
    const float rstd = rsqrtf(red[0] / (float)D + 1e-5f);

    for (int i = tid; i < D; i += 256)
        yr[i] = tobf((xr[i] - mean) * rstd * g[i] + b[i]);
}

__global__ __launch_bounds__(256)
void softmax_causal_k(bf16* __restrict__ scores, int T)
{
    const int t  = blockIdx.x;
    const int hz = blockIdx.y;
    bf16* row = scores + ((long long)hz * T + t) * (long long)T;
    const int tid = threadIdx.x;
    const int n = t + 1;                                   // causal valid length
    __shared__ float red[256];

    float m = -INFINITY;
    for (int u = tid; u < n; u += 256) m = fmaxf(m, (float)row[u]);
    red[tid] = m; __syncthreads();
    for (int o = 128; o > 0; o >>= 1) { if (tid < o) red[tid] = fmaxf(red[tid], red[tid + o]); __syncthreads(); }
    m = red[0]; __syncthreads();

    float s = 0.0f;
    for (int u = tid; u < n; u += 256) { const float e = __expf((float)row[u] - m); row[u] = tobf(e); s += e; }
    red[tid] = s; __syncthreads();
    for (int o = 128; o > 0; o >>= 1) { if (tid < o) red[tid] += red[tid + o]; __syncthreads(); }
    const float inv = 1.0f / red[0];

    for (int u = tid; u < n; u += 256) row[u] = tobf((float)row[u] * inv);
    for (int u = n + tid; u < T; u += 256) row[u] = tobf(0.0f);   // clean tail for PV GEMM
}

__global__ __launch_bounds__(256)
void addheads_k(float* __restrict__ x, const float* __restrict__ att, int T, int D, int HS)
{
    const long long i = (long long)blockIdx.x * 256 + threadIdx.x;
    const int t = (int)(i / D);
    const int c = (int)(i % D);
    const int h = c / HS;
    const int s = c % HS;
    x[i] += att[((long long)h * T + t) * HS + s];
}

__global__ __launch_bounds__(256)
void addinplace_k(float* __restrict__ x, const float* __restrict__ y)
{
    const long long i = (long long)blockIdx.x * 256 + threadIdx.x;
    x[i] += y[i];
}

// ---------------------------------------------------------------------------
// Orchestration
// ---------------------------------------------------------------------------
extern "C" void kernel_launch(void* const* d_in, const int* in_sizes, int n_in,
                              void* d_out, int out_size, void* d_ws, size_t ws_size,
                              hipStream_t stream)
{
    (void)in_sizes; (void)n_in; (void)out_size; (void)ws_size;
    constexpr int T = 2048, D = 1024, H = 16, HS = 64, FF = 4096, NV = 32000, L = 2;

    const int*   inp  = (const int*)  d_in[0];
    const float* tok  = (const float*)d_in[1];
    const float* pos  = (const float*)d_in[2];
    const float* ln1g = (const float*)d_in[3];
    const float* ln1b = (const float*)d_in[4];
    const float* wq   = (const float*)d_in[5];
    const float* wk   = (const float*)d_in[6];
    const float* wv   = (const float*)d_in[7];
    const float* ln2g = (const float*)d_in[8];
    const float* ln2b = (const float*)d_in[9];
    const float* w1   = (const float*)d_in[10];
    const float* b1   = (const float*)d_in[11];
    const float* w2   = (const float*)d_in[12];
    const float* b2   = (const float*)d_in[13];
    const float* lnfg = (const float*)d_in[14];
    const float* lnfb = (const float*)d_in[15];
    const float* wun  = (const float*)d_in[16];
    const float* bun  = (const float*)d_in[17];
    float* out = (float*)d_out;

    const size_t TD = (size_t)T * D;          // 2,097,152

    // ---- workspace carve-up (~205 MB) ----
    char* p = (char*)d_ws;
    float* x    = (float*)p; p += TD * 4;                   // residual stream
    float* h    = (float*)p; p += TD * 4;                   // f32 temp (ffn2 out)
    float* att  = (float*)p; p += TD * 4;                   // [H][T][HS] f32
    bf16*  hb   = (bf16*)p;  p += TD * 2;                   // LN output, bf16
    bf16*  qb   = (bf16*)p;  p += TD * 2;                   // [H][T][HS]
    bf16*  kbb  = (bf16*)p;  p += TD * 2;
    bf16*  vbb  = (bf16*)p;  p += TD * 2;
    bf16*  ffb  = (bf16*)p;  p += (size_t)T * FF * 2;       // [T][FF]
    bf16*  scb  = (bf16*)p;  p += (size_t)4 * T * T * 2;    // [4][T][T] scores/probs
    bf16*  wqb  = (bf16*)p;  p += (size_t)L * H * D * HS * 2;
    bf16*  wkbw = (bf16*)p;  p += (size_t)L * H * D * HS * 2;
    bf16*  wvbw = (bf16*)p;  p += (size_t)L * H * D * HS * 2;
    bf16*  w1b  = (bf16*)p;  p += (size_t)L * D * FF * 2;
    bf16*  w2b  = (bf16*)p;  p += (size_t)L * FF * D * 2;
    bf16*  wunb = (bf16*)p;  p += (size_t)D * NV * 2;

    const dim3 blk(256);
    const long long sHead = (long long)T * HS;

    // ---- one-time (per call) weight conversion f32 -> bf16 ----
    cvt_bf16_k<<<(L * H * D * HS) / 2048, blk, 0, stream>>>(wq, wqb);
    cvt_bf16_k<<<(L * H * D * HS) / 2048, blk, 0, stream>>>(wk, wkbw);
    cvt_bf16_k<<<(L * H * D * HS) / 2048, blk, 0, stream>>>(wv, wvbw);
    cvt_bf16_k<<<(L * D * FF)     / 2048, blk, 0, stream>>>(w1, w1b);
    cvt_bf16_k<<<(L * FF * D)     / 2048, blk, 0, stream>>>(w2, w2b);
    cvt_bf16_k<<<(D * NV)         / 2048, blk, 0, stream>>>(wun, wunb);

    embed_k<<<(T * D) / 256, blk, 0, stream>>>(inp, tok, pos, x, D);

    for (int l = 0; l < L; ++l) {
        ln_k<<<T, blk, 0, stream>>>(x, ln1g + (size_t)l * D, ln1b + (size_t)l * D, hb, D);

        // QKV projections: per-head [T,D] x [D,HS] -> bf16, z = head
        {
            const dim3 g(HS / BN, T / BM, H);               // (1, 16, 16)
            const long long sB = (long long)D * HS;
            const size_t lw = (size_t)l * H * D * HS;
            gemm_wmma_k<<<g, blk, 0, stream>>>(hb, wqb + lw, nullptr, nullptr, qb,
                D, D, HS, HS, 0, sB, sHead, 0, 1.0f, 0, 0, 0);
            gemm_wmma_k<<<g, blk, 0, stream>>>(hb, wkbw + lw, nullptr, nullptr, kbb,
                D, D, HS, HS, 0, sB, sHead, 0, 1.0f, 0, 0, 0);
            gemm_wmma_k<<<g, blk, 0, stream>>>(hb, wvbw + lw, nullptr, nullptr, vbb,
                D, D, HS, HS, 0, sB, sHead, 0, 1.0f, 0, 0, 0);
        }

        // attention, 4 heads per pass (score scratch = 4*T*T bf16 = 32 MB)
        for (int hc = 0; hc < H; hc += 4) {
            bf16* qh = qb  + (size_t)hc * T * HS;
            bf16* kh = kbb + (size_t)hc * T * HS;
            bf16* vh = vbb + (size_t)hc * T * HS;
            float* ah = att + (size_t)hc * T * HS;

            // scores = (Q x K^T) * D^-1/2  (A x B^T, causal block-skip, bf16 out)
            const dim3 gs(T / BN, T / BM, 4);               // (32, 16, 4)
            gemm_wmma_k<<<gs, blk, 0, stream>>>(qh, kh, nullptr, nullptr, scb,
                HS, HS, HS, T, sHead, sHead, (long long)T * T, 0,
                0.03125f /* 1024^-0.5 */, 1 /*transB*/, 0, 1 /*causal*/);

            const dim3 gsm(T, 4);
            softmax_causal_k<<<gsm, blk, 0, stream>>>(scb, T);

            // att_h = P x V  (f32 out)
            const dim3 gp(HS / BN, T / BM, 4);              // (1, 16, 4)
            gemm_wmma_k<<<gp, blk, 0, stream>>>(scb, vh, nullptr, ah, nullptr,
                T, T, HS, HS, (long long)T * T, sHead, sHead, 0,
                1.0f, 0, 0, 0);
        }
        addheads_k<<<(T * D) / 256, blk, 0, stream>>>(x, att, T, D, HS);

        // FFN
        ln_k<<<T, blk, 0, stream>>>(x, ln2g + (size_t)l * D, ln2b + (size_t)l * D, hb, D);
        {
            const dim3 g1(FF / BN, T / BM, 1);              // (64, 16)
            gemm_wmma_k<<<g1, blk, 0, stream>>>(hb, w1b + (size_t)l * D * FF,
                b1 + (size_t)l * FF, nullptr, ffb,
                D, D, FF, FF, 0, 0, 0, 0, 1.0f, 0, 1 /*relu*/, 0);
            const dim3 g2(D / BN, T / BM, 1);               // (16, 16)
            gemm_wmma_k<<<g2, blk, 0, stream>>>(ffb, w2b + (size_t)l * FF * D,
                b2 + (size_t)l * D, h, nullptr,
                FF, FF, D, D, 0, 0, 0, 0, 1.0f, 0, 0, 0);
        }
        addinplace_k<<<(T * D) / 256, blk, 0, stream>>>(x, h);
    }

    // final LN + unembedding [T,D] x [D,V] + b_un  (the 134 GF GEMM)
    ln_k<<<T, blk, 0, stream>>>(x, lnfg, lnfb, hb, D);
    {
        const dim3 g(NV / BN, T / BM, 1);                   // (500, 16)
        gemm_wmma_k<<<g, blk, 0, stream>>>(hb, wunb, bun, out, nullptr,
            D, D, NV, NV, 0, 0, 0, 0, 1.0f, 0, 0, 0);
    }
}
